// ColorGNN_37108517437617
// MI455X (gfx1250) — compile-verified, matrix operand order checked by reading, not run on hardware
//
#include <hip/hip_runtime.h>
#include <hip/hip_bf16.h>

typedef __attribute__((ext_vector_type(16))) _Float16 v16h;
typedef __attribute__((ext_vector_type(8)))  _Float16 v8h;
typedef __attribute__((ext_vector_type(8)))  float    v8f;

#define NNODES 10000
#define MPAD   10016   // padded row count: 10016 = 313 * 32 (GEMM strips are 32 rows)
#define NEDGES 160000
#define MT 2           // 16-row tiles per wave (M)
#define NT 4           // 16-col tiles per wave (N)

__device__ __forceinline__ void atomAddF(float* p, float v) {
  (void)__hip_atomic_fetch_add(p, v, __ATOMIC_RELAXED, __HIP_MEMORY_SCOPE_AGENT);
}

// ---------------- elementwise / prep kernels ----------------

__global__ void k_f32_to_f16(const float* __restrict__ in, _Float16* __restrict__ out, size_t n) {
  size_t i = (size_t)blockIdx.x * blockDim.x + threadIdx.x;
  if (i < n) out[i] = (_Float16)in[i];
}

// W [K x F] f32 row-major  ->  Wt [F x K] f16 row-major
template <int K, int F>
__global__ void k_transpose_to_f16(const float* __restrict__ W, _Float16* __restrict__ Wt) {
  size_t i = (size_t)blockIdx.x * blockDim.x + threadIdx.x;
  if (i >= (size_t)K * F) return;
  int k = (int)(i / F), f = (int)(i % F);   // F is a power of two -> shift/mask
  Wt[(size_t)f * K + k] = (_Float16)W[i];
}

__global__ void k_deg_init(float* __restrict__ deg, int n) {
  int i = blockIdx.x * blockDim.x + threadIdx.x;
  if (i < n) deg[i] = 1.0f;  // self-loop
}

__global__ void k_deg_count(const long long* __restrict__ dst, float* __restrict__ deg, int E) {
  int e = blockIdx.x * blockDim.x + threadIdx.x;
  if (e < E) atomAddF(&deg[dst[e]], 1.0f);
}

__global__ void k_deg_rsqrt(float* __restrict__ deg, int n) {
  int i = blockIdx.x * blockDim.x + threadIdx.x;
  if (i < n) deg[i] = rsqrtf(deg[i]);  // deg >= 1 always (self-loops)
}

// ---------------- WMMA GEMM: C[MPAD x Nc] = A[MPAD x K] * Bt[Nc x K]^T ----------------
// One wave computes a 32x64 output strip: MT x NT = 2x4 tiles of 16x16.
// Per k-step: 4 A b128-loads + 8 B b128-loads feed 8 v_wmma_f32_16x16x32_f16
// (each A fragment reused across 4 WMMAs, each B fragment across 2).
// K and Nc are compile-time so all strides fold into immediate offsets.
// Fragment layout per ISA: lanes 0-15 hold row (tile_base + l), K octets {0..7,16..23};
// lanes 16-31 hold K octets {8..15, 24..31}.

template <int K, int Nc>
__global__ void k_wmma_gemm(const _Float16* __restrict__ A,   // [MPAD x K] row-major
                            const _Float16* __restrict__ Bt,  // [Nc x K] row-major
                            float* __restrict__ C) {          // [MPAD x Nc] row-major
  constexpr int nstrips = Nc / 64;                 // 64 cols per strip
  constexpr int mstrips = MPAD / 32;               // 32 rows per strip
  const int lane = threadIdx.x & 31;
  const int wave = (int)(((size_t)blockIdx.x * blockDim.x + threadIdx.x) >> 5);
  if (wave >= mstrips * nstrips) return;           // wave-uniform guard (EXEC stays all-1s)
  const int ms = wave / nstrips;
  const int ns = wave - ms * nstrips;
  const int half = lane >> 4;
  const int l    = lane & 15;

  const _Float16* abase = A  + (size_t)(ms * 32 + l) * K + half * 8;
  const _Float16* bbase = Bt + (size_t)(ns * 64 + l) * K + half * 8;

  v8f acc[MT][NT] = {};
#pragma unroll 4
  for (int k = 0; k < K; k += 32) {
    v16h a[MT], b[NT];
#pragma unroll
    for (int i = 0; i < MT; ++i) {
      const _Float16* p = abase + i * 16 * K + k;
      v8h lo = *(const v8h*)p;
      v8h hi = *(const v8h*)(p + 16);
#pragma unroll
      for (int j = 0; j < 8; ++j) { a[i][j] = lo[j]; a[i][j + 8] = hi[j]; }
    }
#pragma unroll
    for (int i = 0; i < NT; ++i) {
      const _Float16* p = bbase + i * 16 * K + k;
      v8h lo = *(const v8h*)p;
      v8h hi = *(const v8h*)(p + 16);
#pragma unroll
      for (int j = 0; j < 8; ++j) { b[i][j] = lo[j]; b[i][j + 8] = hi[j]; }
    }
#pragma unroll
    for (int i = 0; i < MT; ++i)
#pragma unroll
      for (int j = 0; j < NT; ++j)
        acc[i][j] = __builtin_amdgcn_wmma_f32_16x16x32_f16(
            false, a[i], false, b[j], (short)0, acc[i][j], false, false);
  }

  // C/D layout: VGPR r -> row (tile_row + half*8 + r), col (tile_col + l)
  float* cbase = C + (size_t)(ms * 32 + half * 8) * Nc + (ns * 64 + l);
#pragma unroll
  for (int i = 0; i < MT; ++i)
#pragma unroll
    for (int j = 0; j < NT; ++j) {
      float* crow = cbase + i * 16 * Nc + j * 16;
#pragma unroll
      for (int r = 0; r < 8; ++r) crow[r * Nc] = acc[i][j][r];  // immediate offsets
    }
}

// ---------------- aggregation kernels ----------------

// OUT[i,f] = XW[i,f] * dinv[i]^2   (self-loop message)
template <int F>
__global__ void k_selfloop(const float* __restrict__ XW, const float* __restrict__ dinv,
                           float* __restrict__ OUT) {
  size_t i = (size_t)blockIdx.x * blockDim.x + threadIdx.x;
  if (i >= (size_t)NNODES * F) return;
  int node = (int)(i / F);                 // power-of-two -> shift
  float di = dinv[node];
  OUT[i] = XW[i] * di * di;
}

// OUT[d] += XW[s] * dinv[s]*dinv[d] ; one thread per (edge, 4-feature chunk)
template <int F>
__global__ void k_scatter(const float* __restrict__ XW,
                          const long long* __restrict__ src,
                          const long long* __restrict__ dst,
                          const float* __restrict__ dinv,
                          float* __restrict__ OUT) {
  constexpr int chunks = F / 4;
  size_t t = (size_t)blockIdx.x * blockDim.x + threadIdx.x;
  if (t >= (size_t)NEDGES * chunks) return;
  int e = (int)(t / chunks);               // power-of-two -> shift
  int c = (int)(t & (chunks - 1)) << 2;
  long long s = src[e], d = dst[e];
  float w = dinv[s] * dinv[d];
  const float4 v = *(const float4*)(XW + (size_t)s * F + c);   // b128 gather
  float* o = OUT + (size_t)d * F + c;
  atomAddF(o + 0, v.x * w);
  atomAddF(o + 1, v.y * w);
  atomAddF(o + 2, v.z * w);
  atomAddF(o + 3, v.w * w);
}

// H[i,f] = f16(relu(OUT[i,f] + b[f]))  -- produces next layer's f16 GEMM input
template <int F>
__global__ void k_bias_relu_f16(const float* __restrict__ OUT, const float* __restrict__ b,
                                _Float16* __restrict__ H) {
  size_t i = (size_t)blockIdx.x * blockDim.x + threadIdx.x;
  if (i >= (size_t)NNODES * F) return;
  int f = (int)(i & (F - 1));
  float v = OUT[i] + b[f];
  H[i] = (_Float16)fmaxf(v, 0.0f);
}

// out[i, 0..2] = H3[i,:] @ Wc + bc   (64 -> 3, trivial VALU work)
__global__ void k_classifier(const _Float16* __restrict__ H, const float* __restrict__ Wc,
                             const float* __restrict__ bc, float* __restrict__ out, int n) {
  int i = blockIdx.x * blockDim.x + threadIdx.x;
  if (i >= n) return;
  float a0 = bc[0], a1 = bc[1], a2 = bc[2];
  const _Float16* h = H + (size_t)i * 64;
#pragma unroll 8
  for (int f = 0; f < 64; ++f) {
    float hv = (float)h[f];
    a0 += hv * Wc[f * 3 + 0];
    a1 += hv * Wc[f * 3 + 1];
    a2 += hv * Wc[f * 3 + 2];
  }
  out[i * 3 + 0] = a0;
  out[i * 3 + 1] = a1;
  out[i * 3 + 2] = a2;
}

// ---------------- host launch ----------------

static inline unsigned blk(size_t n, int b) { return (unsigned)((n + b - 1) / b); }

template <int K, int F>
static void run_layer(const _Float16* Hh, _Float16* Wt, float* XW, float* OUT,
                      const float* dinv, const long long* src, const long long* dst,
                      const float* W, const float* bias, _Float16* Hnext,
                      hipStream_t stream) {
  const int B = 256;
  k_transpose_to_f16<K, F><<<blk((size_t)K * F, B), B, 0, stream>>>(W, Wt);
  constexpr int strips = (MPAD / 32) * (F / 64);        // one wave per 32x64 strip
  k_wmma_gemm<K, F><<<(strips + 7) / 8, B, 0, stream>>>(Hh, Wt, XW);
  size_t nf = (size_t)NNODES * F;                       // downstream ops ignore pad rows
  k_selfloop<F><<<blk(nf, B), B, 0, stream>>>(XW, dinv, OUT);
  k_scatter<F><<<blk((size_t)NEDGES * (F / 4), B), B, 0, stream>>>(XW, src, dst, dinv, OUT);
  k_bias_relu_f16<F><<<blk(nf, B), B, 0, stream>>>(OUT, bias, Hnext);
}

extern "C" void kernel_launch(void* const* d_in, const int* in_sizes, int n_in,
                              void* d_out, int out_size, void* d_ws, size_t ws_size,
                              hipStream_t stream) {
  const float* x  = (const float*)d_in[0];
  const float* W1 = (const float*)d_in[1]; const float* b1 = (const float*)d_in[2];
  const float* W2 = (const float*)d_in[3]; const float* b2 = (const float*)d_in[4];
  const float* W3 = (const float*)d_in[5]; const float* b3 = (const float*)d_in[6];
  const float* Wc = (const float*)d_in[7]; const float* bc = (const float*)d_in[8];
  const long long* edge = (const long long*)d_in[9];
  const long long* src = edge;
  const long long* dst = edge + NEDGES;

  // workspace partition (~52 MB; matrices padded to MPAD rows for 32-row GEMM strips)
  char* ws = (char*)d_ws;
  size_t off = 0;
  auto alloc = [&](size_t bytes) -> void* {
    void* p = ws + off;
    off = (off + bytes + 255) & ~(size_t)255;
    return p;
  };
  float*    dinv = (float*)   alloc((size_t)NNODES * 4);
  _Float16* Hh   = (_Float16*)alloc((size_t)MPAD * 512 * 2);   // f16 GEMM input (reused)
  float*    XW   = (float*)   alloc((size_t)MPAD * 512 * 4);   // GEMM output
  float*    OUT  = (float*)   alloc((size_t)NNODES * 512 * 4); // aggregation accumulator
  _Float16* Wt   = (_Float16*)alloc((size_t)512 * 512 * 2);    // transposed f16 weights

  const int B = 256;

  // degree normalization: deg = 1 + count(dst); dinv = rsqrt(deg)
  k_deg_init <<<blk(NNODES, B), B, 0, stream>>>(dinv, NNODES);
  k_deg_count<<<blk(NEDGES, B), B, 0, stream>>>(dst, dinv, NEDGES);
  k_deg_rsqrt<<<blk(NNODES, B), B, 0, stream>>>(dinv, NNODES);

  // layer 1 input: x -> f16 (pad rows of Hh are never read downstream of GEMM)
  k_f32_to_f16<<<blk((size_t)NNODES * 128, B), B, 0, stream>>>(x, Hh, (size_t)NNODES * 128);

  run_layer<128, 512>(Hh, Wt, XW, OUT, dinv, src, dst, W1, b1, Hh, stream);
  run_layer<512, 256>(Hh, Wt, XW, OUT, dinv, src, dst, W2, b2, Hh, stream);
  run_layer<256, 64 >(Hh, Wt, XW, OUT, dinv, src, dst, W3, b3, Hh, stream);

  k_classifier<<<blk(NNODES, B), B, 0, stream>>>(Hh, Wc, bc, (float*)d_out, NNODES);
}
